// GNNModel_26182120636599
// MI455X (gfx1250) — matrix operand order repaired and checked by reading, not compiled, not run on hardware
//
#include <hip/hip_runtime.h>

#define N_NODES 50000
#define N_EDGES 800000
#define N_GRAPH 64
#define DIN     768
#define HDIM    256
#define ODIM    128
#define BN_EPS  1e-5f

typedef __attribute__((ext_vector_type(16))) __bf16 v16bf;
typedef __attribute__((ext_vector_type(8)))  float  v8f;

// ---------------------------------------------------------------- degrees
__global__ void k_deg_init(float* __restrict__ deg) {
  int i = blockIdx.x * 256 + threadIdx.x;
  if (i < N_NODES) deg[i] = 1.0f;            // self-loop contributes 1
}
__global__ void k_deg_count(const int* __restrict__ dst, float* __restrict__ deg) {
  int i = blockIdx.x * 256 + threadIdx.x;
  if (i < N_EDGES) atomicAdd(&deg[dst[i]], 1.0f);
}
__global__ void k_deg_fin(float* __restrict__ deg) {
  int i = blockIdx.x * 256 + threadIdx.x;
  if (i < N_NODES) deg[i] = rsqrtf(deg[i]);  // deg -> d^-1/2
}

// ------------------------------------------------- pack W into B-fragments
// B layout used here (contiguous k, not the ISA table order; A matches): lane holds col n = lane&15,
// halfs j=0..15 -> k = kt*32 + (lane<16 ? 0 : 16) + j.
// Packed index: ((nt*nKT + kt)*32 + lane)*16 + j. hi/lo bf16 split of fp32.
__global__ void k_pack_w(const float* __restrict__ W, __bf16* __restrict__ hi,
                         __bf16* __restrict__ lo, int K, int Nout) {
  int nKT = K >> 5;
  int tid = blockIdx.x * 256 + threadIdx.x;      // one thread per (nt,kt,lane)
  int total = (Nout >> 4) * nKT * 32;
  if (tid >= total) return;
  int lane = tid & 31;
  int rest = tid >> 5;
  int kt = rest % nKT;
  int nt = rest / nKT;
  int n  = nt * 16 + (lane & 15);
  int kb = kt * 32 + ((lane & 16) ? 16 : 0);
  size_t o = (size_t)tid * 16;
#pragma unroll
  for (int j = 0; j < 16; ++j) {
    float w = W[(size_t)(kb + j) * Nout + n];
    __bf16 h = (__bf16)w;
    hi[o + j] = h;
    lo[o + j] = (__bf16)(w - (float)h);
  }
}

// ------------------------------------------------- GEMM: C = A(fp32) @ W
// bf16x3 split (hi*hi + hi*lo + lo*hi) for ~fp32 accuracy on WMMA.
// Each wave computes 16 rows x 4 N-tiles: one A fragment feeds 12 WMMAs.
// WMMAs are issued pass-major across the 4 independent accumulators so no
// two consecutive WMMAs touch the same D registers (no hazard NOPs).
__global__ __launch_bounds__(256)
void k_gemm(const float* __restrict__ A, const __bf16* __restrict__ Bh,
            const __bf16* __restrict__ Bl, float* __restrict__ C,
            int K, int Nout) {
  const int lane = threadIdx.x & 31;
  const int wave = blockIdx.x * 8 + (threadIdx.x >> 5);
  const int nTN  = Nout >> 4;
  const int wpr  = nTN >> 2;                 // waves per 16-row block (4 tiles each)
  const int tM   = wave / wpr;
  const int tN4  = (wave - tM * wpr) << 2;   // first of 4 N-tiles
  if (tM * 16 >= N_NODES) return;            // wave-uniform guard
  const int nKT  = K >> 5;
  const int row  = tM * 16 + (lane & 15);
  const int koff = (lane & 16) ? 16 : 0;     // A k order (contiguous, as packed B): lanes>=16 start at K+16
  const float* arow = A + (size_t)row * K;
  const size_t tStride = (size_t)nKT * 512;  // halfs per N-tile (nKT*32*16)
  const __bf16* bh = Bh + (size_t)tN4 * tStride + (size_t)lane * 16;
  const __bf16* bl = Bl + (size_t)tN4 * tStride + (size_t)lane * 16;

  const v8f z = {0.f, 0.f, 0.f, 0.f, 0.f, 0.f, 0.f, 0.f};
  v8f acc[4] = {z, z, z, z};
  for (int kt = 0; kt < nKT; ++kt) {
    const int kb = kt * 32 + koff;
    float a[16] __attribute__((aligned(16)));
    *(float4*)(a + 0)  = *(const float4*)(arow + kb);        // K kb..kb+3
    *(float4*)(a + 4)  = *(const float4*)(arow + kb + 4);    // K kb+4..kb+7
    *(float4*)(a + 8)  = *(const float4*)(arow + kb + 8);    // K kb+8..kb+11
    *(float4*)(a + 12) = *(const float4*)(arow + kb + 12);   // K kb+12..kb+15
    v16bf ah, al;
#pragma unroll
    for (int j = 0; j < 16; ++j) {
      __bf16 h = (__bf16)a[j];
      ah[j] = h;
      al[j] = (__bf16)(a[j] - (float)h);
    }
    const __bf16* bhk = bh + (size_t)kt * 512;
    const __bf16* blk = bl + (size_t)kt * 512;
    v16bf wh[4], wl[4];
#pragma unroll
    for (int t = 0; t < 4; ++t) {
      wh[t] = *(const v16bf*)(bhk + (size_t)t * tStride);
      wl[t] = *(const v16bf*)(blk + (size_t)t * tStride);
    }
    // pass 1: hi(A) * hi(B) -- 4 independent accumulators back-to-back
#pragma unroll
    for (int t = 0; t < 4; ++t)
      acc[t] = __builtin_amdgcn_wmma_f32_16x16x32_bf16(false, ah, false, wh[t],
                                                       (short)0, acc[t], false, false);
    // pass 2: hi(A) * lo(B)
#pragma unroll
    for (int t = 0; t < 4; ++t)
      acc[t] = __builtin_amdgcn_wmma_f32_16x16x32_bf16(false, ah, false, wl[t],
                                                       (short)0, acc[t], false, false);
    // pass 3: lo(A) * hi(B)
#pragma unroll
    for (int t = 0; t < 4; ++t)
      acc[t] = __builtin_amdgcn_wmma_f32_16x16x32_bf16(false, al, false, wh[t],
                                                       (short)0, acc[t], false, false);
  }
  // D layout: VGPR r -> row r (lanes 0-15) / r+8 (lanes 16-31), col = lane&15
  const int rbase = tM * 16 + ((lane & 16) ? 8 : 0);
#pragma unroll
  for (int t = 0; t < 4; ++t) {
    const int col = (tN4 + t) * 16 + (lane & 15);
    float* cp = C + (size_t)rbase * Nout + col;
#pragma unroll
    for (int r = 0; r < 8; ++r) cp[(size_t)r * Nout] = acc[t][r];
  }
}

// --------------------------------------------- edge aggregation (scatter)
__global__ void k_agg(const int* __restrict__ src, const int* __restrict__ dst,
                      const float* __restrict__ dinv, const float* __restrict__ h,
                      float* __restrict__ agg, int F, int tpeShift) {
  long tid = (long)blockIdx.x * 256 + threadIdx.x;
  int e = (int)(tid >> tpeShift);
  int c = ((int)tid & ((1 << tpeShift) - 1)) << 2;
  if (e >= N_EDGES) return;
  int s = src[e], d = dst[e];
  float coef = dinv[s] * dinv[d];
  const float4 v = *(const float4*)(h + (size_t)s * F + c);
  float* p = agg + (size_t)d * F + c;
  atomicAdd(p + 0, v.x * coef);
  atomicAdd(p + 1, v.y * coef);
  atomicAdd(p + 2, v.z * coef);
  atomicAdd(p + 3, v.w * coef);
}

// -------------------------- self-loop + bias + BN(eval) + ReLU, in place
__global__ void k_pw_bn_relu(float* __restrict__ agg, const float* __restrict__ h,
                             const float* __restrict__ dinv,
                             const float* __restrict__ b, const float* __restrict__ g,
                             const float* __restrict__ be, const float* __restrict__ rm,
                             const float* __restrict__ rv) {
  long t = (long)blockIdx.x * 256 + threadIdx.x;
  if (t >= (long)N_NODES * HDIM) return;
  int tid = (int)t;
  int n = tid >> 8;          // HDIM == 256
  int f = tid & 255;
  float di = dinv[n];
  float v = agg[tid] + h[tid] * di * di + b[f];
  v = (v - rm[f]) * rsqrtf(rv[f] + BN_EPS) * g[f] + be[f];
  agg[tid] = fmaxf(v, 0.0f);
}

// ------------------------------------------------- pooling pieces
__global__ void k_cnt(const int* __restrict__ batch, float* __restrict__ cnt) {
  int i = blockIdx.x * 256 + threadIdx.x;
  if (i < N_NODES) atomicAdd(&cnt[batch[i]], 1.0f);
}
__global__ void k_pw3_pool(const float* __restrict__ agg, const float* __restrict__ h,
                           const float* __restrict__ dinv, const float* __restrict__ b,
                           const int* __restrict__ batch, float* __restrict__ pool) {
  long t = (long)blockIdx.x * 256 + threadIdx.x;
  if (t >= (long)N_NODES * ODIM) return;
  int tid = (int)t;
  int n = tid >> 7;          // ODIM == 128
  int f = tid & 127;
  float di = dinv[n];
  float v = agg[tid] + h[tid] * di * di + b[f];
  atomicAdd(&pool[batch[n] * ODIM + f], v);
}
__global__ void k_final(const float* __restrict__ pool, const float* __restrict__ cnt,
                        float* __restrict__ out) {
  int tid = blockIdx.x * 256 + threadIdx.x;
  if (tid >= N_GRAPH * ODIM) return;
  out[tid] = pool[tid] / fmaxf(cnt[tid >> 7], 1.0f);
}

// ----------------------------------------------------------------- launch
extern "C" void kernel_launch(void* const* d_in, const int* in_sizes, int n_in,
                              void* d_out, int out_size, void* d_ws, size_t ws_size,
                              hipStream_t stream) {
  (void)in_sizes; (void)n_in; (void)out_size; (void)ws_size;
  const float* x   = (const float*)d_in[0];
  const int*   ei  = (const int*)d_in[1];
  const int*   bat = (const int*)d_in[2];
  const float* W1  = (const float*)d_in[3];
  const float* b1  = (const float*)d_in[4];
  const float* g1  = (const float*)d_in[5];
  const float* be1 = (const float*)d_in[6];
  const float* rm1 = (const float*)d_in[7];
  const float* rv1 = (const float*)d_in[8];
  const float* W2  = (const float*)d_in[9];
  const float* b2  = (const float*)d_in[10];
  const float* g2  = (const float*)d_in[11];
  const float* be2 = (const float*)d_in[12];
  const float* rm2 = (const float*)d_in[13];
  const float* rv2 = (const float*)d_in[14];
  const float* W3  = (const float*)d_in[15];
  const float* b3  = (const float*)d_in[16];
  const int* src = ei;
  const int* dst = ei + N_EDGES;

  char* ws = (char*)d_ws;
  size_t off = 0;
  auto take = [&](size_t bytes) -> char* {
    char* p = ws + off;
    off = (off + bytes + 255) & ~(size_t)255;
    return p;
  };
  float*  dinv = (float*)take((size_t)N_NODES * 4);
  __bf16* p1h  = (__bf16*)take((size_t)DIN * HDIM * 2);
  __bf16* p1l  = (__bf16*)take((size_t)DIN * HDIM * 2);
  __bf16* p2h  = (__bf16*)take((size_t)HDIM * HDIM * 2);
  __bf16* p2l  = (__bf16*)take((size_t)HDIM * HDIM * 2);
  __bf16* p3h  = (__bf16*)take((size_t)HDIM * ODIM * 2);
  __bf16* p3l  = (__bf16*)take((size_t)HDIM * ODIM * 2);
  float* bufA  = (float*)take((size_t)N_NODES * HDIM * 4);  // h1 -> h2 -> agg3
  float* bufB  = (float*)take((size_t)N_NODES * HDIM * 4);  // agg1/x2 -> h3
  float* bufC  = (float*)take((size_t)N_NODES * HDIM * 4);  // agg2/x3
  float* pool  = (float*)take((size_t)N_GRAPH * ODIM * 4);
  float* cnt   = (float*)take((size_t)N_GRAPH * 4);

  const unsigned NB = (N_NODES + 255) / 256;
  const unsigned EB = (N_EDGES + 255) / 256;

  // degrees -> d^-1/2
  k_deg_init<<<NB, 256, 0, stream>>>(dinv);
  k_deg_count<<<EB, 256, 0, stream>>>(dst, dinv);
  k_deg_fin<<<NB, 256, 0, stream>>>(dinv);

  // pack weights into WMMA B-fragment order (bf16 hi/lo)
  k_pack_w<<<((HDIM/16)*(DIN/32)*32 + 255)/256, 256, 0, stream>>>(W1, p1h, p1l, DIN, HDIM);
  k_pack_w<<<((HDIM/16)*(HDIM/32)*32 + 255)/256, 256, 0, stream>>>(W2, p2h, p2l, HDIM, HDIM);
  k_pack_w<<<((ODIM/16)*(HDIM/32)*32 + 255)/256, 256, 0, stream>>>(W3, p3h, p3l, HDIM, ODIM);

  // GEMM grids: (rows/16) * (Ntiles/4) waves, 8 waves per block
  const unsigned G12 = ((N_NODES/16) * (HDIM/16/4) + 7) / 8;  // 12500 waves
  const unsigned G3  = ((N_NODES/16) * (ODIM/16/4) + 7) / 8;  //  6250 waves

  // ---- layer 1 ----
  k_gemm<<<G12, 256, 0, stream>>>(x, p1h, p1l, bufA, DIN, HDIM);
  hipMemsetAsync(bufB, 0, (size_t)N_NODES * HDIM * 4, stream);
  k_agg<<<(unsigned)(((size_t)N_EDGES*(HDIM/4) + 255)/256), 256, 0, stream>>>(src, dst, dinv, bufA, bufB, HDIM, 6);
  k_pw_bn_relu<<<(unsigned)(((size_t)N_NODES*HDIM + 255)/256), 256, 0, stream>>>(bufB, bufA, dinv, b1, g1, be1, rm1, rv1);

  // ---- layer 2 ----
  k_gemm<<<G12, 256, 0, stream>>>(bufB, p2h, p2l, bufA, HDIM, HDIM);
  hipMemsetAsync(bufC, 0, (size_t)N_NODES * HDIM * 4, stream);
  k_agg<<<(unsigned)(((size_t)N_EDGES*(HDIM/4) + 255)/256), 256, 0, stream>>>(src, dst, dinv, bufA, bufC, HDIM, 6);
  k_pw_bn_relu<<<(unsigned)(((size_t)N_NODES*HDIM + 255)/256), 256, 0, stream>>>(bufC, bufA, dinv, b2, g2, be2, rm2, rv2);

  // ---- layer 3 + pool ----
  k_gemm<<<G3, 256, 0, stream>>>(bufC, p3h, p3l, bufB, HDIM, ODIM);
  hipMemsetAsync(bufA, 0, (size_t)N_NODES * ODIM * 4, stream);
  k_agg<<<(unsigned)(((size_t)N_EDGES*(ODIM/4) + 255)/256), 256, 0, stream>>>(src, dst, dinv, bufB, bufA, ODIM, 5);
  hipMemsetAsync(pool, 0, (size_t)N_GRAPH * ODIM * 4, stream);
  hipMemsetAsync(cnt, 0, (size_t)N_GRAPH * 4, stream);
  k_cnt<<<NB, 256, 0, stream>>>(bat, cnt);
  k_pw3_pool<<<(unsigned)(((size_t)N_NODES*ODIM + 255)/256), 256, 0, stream>>>(bufA, bufB, dinv, b3, bat, pool);
  k_final<<<(N_GRAPH*ODIM + 255)/256, 256, 0, stream>>>(pool, cnt, (float*)d_out);
}
